// IronRoPESelfAttention_67860483276956
// MI455X (gfx1250) — matrix operand order repaired and checked by reading, not compiled, not run on hardware
//
#include <hip/hip_runtime.h>
#include <hip/hip_bf16.h>

// Problem constants (from reference)
#define B_   2
#define T_   1024
#define C_   1024
#define H_   16
#define D_   64
#define RM_  32
#define FBM_ 32

typedef __attribute__((ext_vector_type(16))) __bf16 v16bf;
typedef __attribute__((ext_vector_type(2)))  __bf16 v2bf;
typedef __attribute__((ext_vector_type(8)))  float  v8f;
typedef __attribute__((ext_vector_type(4)))  int    v4i;

union F8    { v8f  v; float f[8]; };
union BF16F { v16bf v; unsigned short h[16]; unsigned int u[8]; uint4 q[2]; };

// ---------------------------------------------------------------------------
// bf16 conversion via native fptrunc (backend selects HW v_cvt for gfx1250;
// gfx1250 has native BF16 VALU support per the ISA)
// ---------------------------------------------------------------------------
__device__ __forceinline__ unsigned short f2bf(float f) {
    return __builtin_bit_cast(unsigned short, (__bf16)f);
}

__device__ __forceinline__ unsigned int pk2bf(float lo, float hi) {
    v2bf p; p[0] = (__bf16)lo; p[1] = (__bf16)hi;
    return __builtin_bit_cast(unsigned int, p);
}

// ---------------------------------------------------------------------------
// Async global->LDS copy (CDNA5 GLOBAL_LOAD_ASYNC_TO_LDS_B128, ASYNCcnt).
// Builtin signature (from compile probe): (v4i AS1*, v4i AS3*, imm, imm).
// Synchronous uint4 fallback if the builtin is not exposed.
// ---------------------------------------------------------------------------
#if __has_builtin(__builtin_amdgcn_global_load_async_to_lds_b128)
#define HAVE_ASYNC_LDS 1
#else
#define HAVE_ASYNC_LDS 0
#endif

typedef __attribute__((address_space(1))) v4i* as1_v4i_p;  // global
typedef __attribute__((address_space(3))) v4i* as3_v4i_p;  // LDS

__device__ __forceinline__ void async_cp16(const unsigned short* g, unsigned short* l) {
#if HAVE_ASYNC_LDS
    __builtin_amdgcn_global_load_async_to_lds_b128(
        (as1_v4i_p)(void*)g, (as3_v4i_p)(void*)l, 0, 0);
#else
    *(uint4*)l = *(const uint4*)g;
#endif
}

#if HAVE_ASYNC_LDS
#  if __has_builtin(__builtin_amdgcn_s_wait_asynccnt)
#    define ASYNC_WAIT(n) __builtin_amdgcn_s_wait_asynccnt(n)
#  else
#    define ASYNC_WAIT(n) asm volatile("s_wait_asynccnt " #n ::: "memory")
#  endif
#else
#  define ASYNC_WAIT(n) ((void)0)
#endif

// ---------------------------------------------------------------------------
// Kernel 1: qkv = x @ W_attn^T + b_attn, fused RoPE on q/k, head-major stores.
//   q,k -> (B,H,T,D) bf16 ; v -> (B,H,D,T) bf16 (transposed for PV B-operand)
// Each wave computes a 16x64 output strip: the A fragment is loaded once per
// K-step and reused by 4 WMMAs against 4 B fragments staged in LDS.
// ---------------------------------------------------------------------------
__global__ __launch_bounds__(256)
void qkv_rope_kernel(const float* __restrict__ x, const float* __restrict__ coords,
                     const float* __restrict__ Wattn, const float* __restrict__ battn,
                     const float* __restrict__ Wrope,
                     unsigned short* __restrict__ qws,
                     unsigned short* __restrict__ kws,
                     unsigned short* __restrict__ vws)
{
    __shared__ __align__(16) unsigned short sW[64 * 32];  // 64 n-cols x 32 K, bf16
    unsigned int* sW32 = (unsigned int*)sW;

    const int lane    = threadIdx.x & 31;
    const int wave    = threadIdx.x >> 5;
    const int laneLow = (lane < 16);
    const int koffA   = laneLow ? 0 : 8;
    const int nbase   = blockIdx.x * 64;                 // feature strip (of 3C)
    const int mbase   = blockIdx.y * 128 + wave * 16;    // token-row tile (of B*T)

    F8 acc[4];
#pragma unroll
    for (int j = 0; j < 4; ++j)
#pragma unroll
        for (int i = 0; i < 8; ++i) acc[j].f[i] = 0.f;

    const int rowA = mbase + (lane & 15);
    const float* xrow = x + (size_t)rowA * C_;

    for (int ks = 0; ks < C_ / 32; ++ks) {
        __syncthreads();
        // Stage W_attn[nbase..+63, ks*32..+31] as packed bf16 pairs.
        // 1024 dword-pairs by 256 threads, 4 each: float2 load + ds_store_b32.
#pragma unroll
        for (int i = 0; i < 4; ++i) {
            const int p  = threadIdx.x + i * 256;  // pair index; row = p>>4
            const int r  = p >> 4;
            const int kk = (p & 15) * 2;
            const float2 w = *(const float2*)(Wattn + (size_t)(nbase + r) * C_ + ks * 32 + kk);
            sW32[p] = pk2bf(w.x, w.y);
        }
        __syncthreads();

        // A fragment: x row fp32 -> bf16 (low lanes K{0..7,16..23}, high {8..15,24..31})
        const float4 fa0 = *(const float4*)(xrow + ks * 32 + koffA);
        const float4 fa1 = *(const float4*)(xrow + ks * 32 + koffA + 4);
        const float4 fb0 = *(const float4*)(xrow + ks * 32 + koffA + 16);
        const float4 fb1 = *(const float4*)(xrow + ks * 32 + koffA + 20);
        BF16F a;
        a.u[0] = pk2bf(fa0.x, fa0.y); a.u[1] = pk2bf(fa0.z, fa0.w);
        a.u[2] = pk2bf(fa1.x, fa1.y); a.u[3] = pk2bf(fa1.z, fa1.w);
        a.u[4] = pk2bf(fb0.x, fb0.y); a.u[5] = pk2bf(fb0.z, fb0.w);
        a.u[6] = pk2bf(fb1.x, fb1.y); a.u[7] = pk2bf(fb1.z, fb1.w);

        if (ks + 1 < C_ / 32)
            __builtin_prefetch(xrow + (ks + 1) * 32 + koffA, 0, 3);  // global_prefetch_b8

#pragma unroll
        for (int j = 0; j < 4; ++j) {
            const unsigned short* bp = sW + (j * 16 + (lane & 15)) * 32 + (laneLow ? 0 : 8);
            BF16F bfr;
            bfr.q[0] = *(const uint4*)(bp);
            bfr.q[1] = *(const uint4*)(bp + 16);
            acc[j].v = __builtin_amdgcn_wmma_f32_16x16x32_bf16(
                           false, a.v, false, bfr.v, (short)0, acc[j].v, false, false);
        }
    }

    // Epilogue: bias, RoPE (q,k), head-major scatter stores
#pragma unroll
    for (int j = 0; j < 4; ++j) {
        const int n    = nbase + j * 16 + (lane & 15);
        const int type = n >> 10;          // 0=q 1=k 2=v
        const int cc   = n & (C_ - 1);
        const int hh   = cc >> 6;
        const int d    = cc & 63;
        const float bn = battn[n];
#pragma unroll
        for (int r = 0; r < 8; ++r) {
            const int rowr = mbase + r + (laneLow ? 0 : 8);   // C-layout row
            const int bb   = rowr >> 10;
            const int t    = rowr & (T_ - 1);
            float val = acc[j].f[r] + bn;
            if (type < 2) {  // uniform across the wave (16-wide subtile)
                float theta = coords[bb * T_ + t] * Wrope[d >> 1];
                float sn, cs; __sincosf(theta, &sn, &cs);
                float part = __shfl_xor(val, 1, 32);  // RoPE pair is the adjacent lane
                val = ((d & 1) == 0) ? (val * cs - part * sn) : (part * sn + val * cs);
                unsigned short* dst = (type == 0) ? qws : kws;
                dst[(((size_t)(bb * H_ + hh)) * T_ + t) * D_ + d] = f2bf(val);
            } else {
                vws[(((size_t)(bb * H_ + hh)) * D_ + d) * T_ + t] = f2bf(val);
            }
        }
    }
}

// ---------------------------------------------------------------------------
// Kernel 2: Fourier relative-position bias table (head independent), 8 MB,
// L2-resident for the flash kernel.
// ---------------------------------------------------------------------------
__global__ __launch_bounds__(256)
void fourier_bias_kernel(const float* __restrict__ coords, const float* __restrict__ Wfb,
                         const float* __restrict__ bcos, const float* __restrict__ bsin,
                         float* __restrict__ biasT)
{
    const int idx = blockIdx.x * 256 + threadIdx.x;     // B*T*T = 2^21 threads
    const int b   = idx >> 20;
    const int rem = idx & ((1 << 20) - 1);
    const int qi  = rem >> 10, ki = rem & 1023;
    const float delta = coords[b * T_ + qi] - coords[b * T_ + ki];
    float acc = 0.f;
#pragma unroll 4
    for (int m = 0; m < FBM_; ++m) {
        float sn, cs; __sincosf(delta * Wfb[m], &sn, &cs);
        acc += cs * bcos[m] + sn * bsin[m];
    }
    biasT[idx] = acc * 0.17677669529663688f;  // 1/sqrt(32)
}

// ---------------------------------------------------------------------------
// Kernel 3: flash attention per (b, h, 64-row q block). 4 waves/block; each
// wave owns 16 q rows. K/V chunks (32 columns) are shared by all 4 waves and
// staged in LDS with double-buffered async copies (GLOBAL_LOAD_ASYNC_TO_LDS),
// cutting global K/V traffic 4x. Per chunk: 4 score WMMAs + online softmax +
// per-wave LDS round-trip (C-layout -> A-layout for P) + 4 PV WMMAs.
// ---------------------------------------------------------------------------
__global__ __launch_bounds__(128)
void flash_attn_kernel(const unsigned short* __restrict__ qws,
                       const unsigned short* __restrict__ kws,
                       const unsigned short* __restrict__ vws,
                       const float* __restrict__ biasT,
                       unsigned short* __restrict__ yh)
{
    __shared__ __align__(16) unsigned short sK[2][32 * 64];  // [buf][col][d]
    __shared__ __align__(16) unsigned short sV[2][64 * 32];  // [buf][d][t]
    __shared__ __align__(16) unsigned short sP[4][16 * 32];  // per-wave P tile

    const int tid     = threadIdx.x;        // 0..127
    const int lane    = tid & 31;
    const int wave    = tid >> 5;
    const int laneLow = (lane < 16);
    const int koffA   = laneLow ? 0 : 8;
    const int koffB   = laneLow ? 0 : 8;
    const int b  = blockIdx.z, hh = blockIdx.y;
    const int qbase = blockIdx.x * 64 + wave * 16;
    const size_t bh = (size_t)(b * H_ + hh);

    // Stage one 32-column K/V chunk into LDS buffer `buf` (4 x 16B per thread).
    auto stage = [&](int buf, int kc) {
        const unsigned short* kg = kws + (bh * T_ + kc) * D_;  // 4 KB contiguous
#pragma unroll
        for (int i = 0; i < 2; ++i) {
            const int o = (tid + i * 128) * 8;                 // ushort offset
            async_cp16(kg + o, &sK[buf][o]);
        }
#pragma unroll
        for (int i = 0; i < 2; ++i) {
            const int piece = tid + i * 128;                   // 64 rows x 4 x 16B
            const int row = piece >> 2;
            const int qo  = (piece & 3) * 8;
            async_cp16(vws + (bh * D_ + row) * T_ + kc + qo, &sV[buf][row * 32 + qo]);
        }
    };

    // Q fragments for K=0..31 and K=32..63 (D = 64)
    const unsigned short* qp = qws + (bh * T_ + qbase + (lane & 15)) * D_;
    BF16F a0, a1;
    a0.q[0] = *(const uint4*)(qp + koffA);
    a0.q[1] = *(const uint4*)(qp + koffA + 16);
    a1.q[0] = *(const uint4*)(qp + 32 + koffA);
    a1.q[1] = *(const uint4*)(qp + 32 + koffA + 16);

    float mrun[8], lrun[8];
    F8 o[4];
#pragma unroll
    for (int r = 0; r < 8; ++r) { mrun[r] = -1e30f; lrun[r] = 0.f; }
#pragma unroll
    for (int dt = 0; dt < 4; ++dt)
#pragma unroll
        for (int r = 0; r < 8; ++r) o[dt].f[r] = 0.f;

    unsigned short* sPw = sP[wave];
    const float* bq = biasT + (size_t)b * T_ * T_;

    stage(0, 0);  // prologue: fill buffer 0

    for (int it = 0; it < T_ / 32; ++it) {
        const int kc  = it * 32;
        const int cur = it & 1;
        if (kc + 32 < T_) {
            stage(cur ^ 1, kc + 32);  // prefetch next chunk into other buffer
            ASYNC_WAIT(4);            // current chunk (first 4 per-wave ops) done
        } else {
            ASYNC_WAIT(0);
        }
        __syncthreads();

        F8 s0, s1;
#pragma unroll
        for (int r = 0; r < 8; ++r) { s0.f[r] = 0.f; s1.f[r] = 0.f; }

#pragma unroll
        for (int j = 0; j < 2; ++j) {  // two 16-column score tiles
            const unsigned short* kp = &sK[cur][(j * 16 + (lane & 15)) * 64];
            BF16F blo, bhi;
            blo.q[0] = *(const uint4*)(kp + koffB);
            blo.q[1] = *(const uint4*)(kp + koffB + 16);
            bhi.q[0] = *(const uint4*)(kp + 32 + koffB);
            bhi.q[1] = *(const uint4*)(kp + 32 + koffB + 16);
            F8& sj = j ? s1 : s0;
            sj.v = __builtin_amdgcn_wmma_f32_16x16x32_bf16(false, a0.v, false, blo.v, (short)0, sj.v, false, false);
            sj.v = __builtin_amdgcn_wmma_f32_16x16x32_bf16(false, a1.v, false, bhi.v, (short)0, sj.v, false, false);
        }

        // scale + bias + online softmax (row stats via half-wave shfl reductions)
#pragma unroll
        for (int r = 0; r < 8; ++r) {
            const int rowr = qbase + r + (laneLow ? 0 : 8);
            const int col  = kc + (lane & 15);
            float v0 = s0.f[r] * 0.125f + bq[(size_t)rowr * T_ + col];       // 1/sqrt(64)
            float v1 = s1.f[r] * 0.125f + bq[(size_t)rowr * T_ + col + 16];
            float mx = fmaxf(v0, v1);
#pragma unroll
            for (int msk = 1; msk < 16; msk <<= 1) mx = fmaxf(mx, __shfl_xor(mx, msk, 32));
            const float mnew = fmaxf(mrun[r], mx);
            const float corr = __expf(mrun[r] - mnew);
            const float p0 = __expf(v0 - mnew);
            const float p1 = __expf(v1 - mnew);
            float rs = p0 + p1;
#pragma unroll
            for (int msk = 1; msk < 16; msk <<= 1) rs += __shfl_xor(rs, msk, 32);
            lrun[r] = lrun[r] * corr + rs;
            mrun[r] = mnew;
#pragma unroll
            for (int dt = 0; dt < 4; ++dt) o[dt].f[r] *= corr;
            const int prow = r + (laneLow ? 0 : 8);
            sPw[prow * 32 + (lane & 15)]      = f2bf(p0);
            sPw[prow * 32 + (lane & 15) + 16] = f2bf(p1);
        }
        __syncthreads();

        // P back out of LDS in WMMA A layout (2x ds_load_b128 per lane)
        BF16F pA;
        const unsigned short* pp = sPw + (lane & 15) * 32 + koffA;
        pA.q[0] = *(const uint4*)(pp);
        pA.q[1] = *(const uint4*)(pp + 16);

#pragma unroll
        for (int dt = 0; dt < 4; ++dt) {
            const unsigned short* vp = &sV[cur][(dt * 16 + (lane & 15)) * 32 + koffB];
            BF16F bv;
            bv.q[0] = *(const uint4*)(vp);
            bv.q[1] = *(const uint4*)(vp + 16);
            o[dt].v = __builtin_amdgcn_wmma_f32_16x16x32_bf16(false, pA.v, false, bv.v, (short)0, o[dt].v, false, false);
        }
        __syncthreads();  // protect sP/sK/sV before next chunk rewrites them
    }

    // normalize and store y head slice (B,T,C) bf16
#pragma unroll
    for (int r = 0; r < 8; ++r) {
        const int rowr = qbase + r + (laneLow ? 0 : 8);
        const float inv = 1.0f / lrun[r];
#pragma unroll
        for (int dt = 0; dt < 4; ++dt) {
            yh[((size_t)b * T_ + rowr) * C_ + hh * D_ + dt * 16 + (lane & 15)] =
                f2bf(o[dt].f[r] * inv);
        }
    }
}

// ---------------------------------------------------------------------------
// Kernel 4: out = x + y @ W_proj^T + b_proj.  Same 16x64-per-wave WMMA tiling
// as kernel 1; A operand (y) is already bf16.
// ---------------------------------------------------------------------------
__global__ __launch_bounds__(256)
void proj_residual_kernel(const unsigned short* __restrict__ yh,
                          const float* __restrict__ Wproj,
                          const float* __restrict__ bproj,
                          const float* __restrict__ x,
                          float* __restrict__ out)
{
    __shared__ __align__(16) unsigned short sW[64 * 32];
    unsigned int* sW32 = (unsigned int*)sW;

    const int lane    = threadIdx.x & 31;
    const int wave    = threadIdx.x >> 5;
    const int laneLow = (lane < 16);
    const int koffA   = laneLow ? 0 : 8;
    const int nbase   = blockIdx.x * 64;
    const int mbase   = blockIdx.y * 128 + wave * 16;

    F8 acc[4];
#pragma unroll
    for (int j = 0; j < 4; ++j)
#pragma unroll
        for (int i = 0; i < 8; ++i) acc[j].f[i] = 0.f;

    const int rowA = mbase + (lane & 15);
    const unsigned short* yrow = yh + (size_t)rowA * C_;

    for (int ks = 0; ks < C_ / 32; ++ks) {
        __syncthreads();
#pragma unroll
        for (int i = 0; i < 4; ++i) {
            const int p  = threadIdx.x + i * 256;
            const int r  = p >> 4;
            const int kk = (p & 15) * 2;
            const float2 w = *(const float2*)(Wproj + (size_t)(nbase + r) * C_ + ks * 32 + kk);
            sW32[p] = pk2bf(w.x, w.y);
        }
        __syncthreads();

        BF16F a;
        a.q[0] = *(const uint4*)(yrow + ks * 32 + koffA);
        a.q[1] = *(const uint4*)(yrow + ks * 32 + koffA + 16);

        if (ks + 1 < C_ / 32)
            __builtin_prefetch(yrow + (ks + 1) * 32 + koffA, 0, 3);

#pragma unroll
        for (int j = 0; j < 4; ++j) {
            const unsigned short* bp = sW + (j * 16 + (lane & 15)) * 32 + (laneLow ? 0 : 8);
            BF16F bfr;
            bfr.q[0] = *(const uint4*)(bp);
            bfr.q[1] = *(const uint4*)(bp + 16);
            acc[j].v = __builtin_amdgcn_wmma_f32_16x16x32_bf16(
                           false, a.v, false, bfr.v, (short)0, acc[j].v, false, false);
        }
    }

#pragma unroll
    for (int j = 0; j < 4; ++j) {
        const int n  = nbase + j * 16 + (lane & 15);
        const float bn = bproj[n];
#pragma unroll
        for (int r = 0; r < 8; ++r) {
            const int rowr = mbase + r + (laneLow ? 0 : 8);
            out[(size_t)rowr * C_ + n] = acc[j].f[r] + bn + x[(size_t)rowr * C_ + n];
        }
    }
}

// ---------------------------------------------------------------------------
extern "C" void kernel_launch(void* const* d_in, const int* in_sizes, int n_in,
                              void* d_out, int out_size, void* d_ws, size_t ws_size,
                              hipStream_t stream)
{
    const float* x      = (const float*)d_in[0];
    const float* coords = (const float*)d_in[1];
    const float* Wattn  = (const float*)d_in[2];
    const float* battn  = (const float*)d_in[3];
    const float* Wproj  = (const float*)d_in[4];
    const float* bproj  = (const float*)d_in[5];
    const float* Wrope  = (const float*)d_in[6];
    const float* Wfb    = (const float*)d_in[7];
    const float* bcos   = (const float*)d_in[8];
    const float* bsin   = (const float*)d_in[9];
    float* out = (float*)d_out;
    (void)in_sizes; (void)n_in; (void)out_size; (void)ws_size;

    // Workspace layout (24 MB total):
    //   [0,4M)   q  (B,H,T,D) bf16
    //   [4,8M)   k  (B,H,T,D) bf16
    //   [8,12M)  vT (B,H,D,T) bf16
    //   [12,20M) bias (B,T,T) fp32
    //   [20,24M) y  (B,T,C) bf16
    char* ws = (char*)d_ws;
    unsigned short* qws = (unsigned short*)(ws);
    unsigned short* kws = (unsigned short*)(ws + (size_t)(4u << 20));
    unsigned short* vws = (unsigned short*)(ws + (size_t)(8u << 20));
    float*          bT  = (float*)        (ws + (size_t)(12u << 20));
    unsigned short* yh  = (unsigned short*)(ws + (size_t)(20u << 20));

    qkv_rope_kernel<<<dim3(3 * C_ / 64, (B_ * T_) / 128), 256, 0, stream>>>(
        x, coords, Wattn, battn, Wrope, qws, kws, vws);

    fourier_bias_kernel<<<dim3((B_ * T_ * T_) / 256), 256, 0, stream>>>(
        coords, Wfb, bcos, bsin, bT);

    flash_attn_kernel<<<dim3(T_ / 64, H_, B_), 128, 0, stream>>>(
        qws, kws, vws, bT, yh);

    proj_residual_kernel<<<dim3(C_ / 64, (B_ * T_) / 128), 256, 0, stream>>>(
        yh, Wproj, bproj, x, out);
}
